// Filter_network_3882650435767
// MI455X (gfx1250) — matrix operand order repaired and checked by reading, not compiled, not run on hardware
//
#include <hip/hip_runtime.h>
#include <hip/hip_bf16.h>

typedef __attribute__((ext_vector_type(16))) _Float16 v16h;
typedef __attribute__((ext_vector_type(8)))  float    v8f;

#define PI_F 3.14159265358979323846f

// Problem constants
#define BS   16
#define N1   64
#define T1   16384      // output length of both convs
#define KF   250        // fixed-filter taps
#define KFP  256        // padded K for WMMA
#define KG   128        // gabor taps
#define PREV 127        // prevFrame length (L1-2)
#define TLEN 16633      // xIn length
#define FB   8193       // F_BINS
#define M2   8192       // complex FFT length (T1/2)

// ---------------------------------------------------------------------------
// K1: Gabor kernels [16,64,128] (f32) + diffed fixed filters dF [64,256] (f16)
// ---------------------------------------------------------------------------
__global__ void prep_kernel(const float* __restrict__ QPrev,         // [16,64]
                            const float* __restrict__ fc,            // [64]
                            const float* __restrict__ filtersFixed,  // [65,250]
                            float* __restrict__ gab,                 // [16,64,128]
                            _Float16* __restrict__ dFh)              // [64,256]
{
    int i = blockIdx.x * 256 + threadIdx.x;
    if (i < BS * N1 * KG) {
        int k = i & (KG - 1);
        int n = (i >> 7) & (N1 - 1);
        int b = i >> 13;
        float t     = ((float)k - (KG - 1) * 0.5f) / 16000.0f;
        float Q     = 1.0f + 10.0f * QPrev[b * N1 + n];
        float phase = 2.0f * PI_F * fc[n] * t;
        float r     = phase / Q;
        gab[i] = cosf(phase) * expf(-0.5f * r * r);
    }
    if (i < N1 * KFP) {
        int k = i & (KFP - 1);
        int c = i >> 8;
        float v = (k < KF) ? (filtersFixed[(c + 1) * KF + k] - filtersFixed[c * KF + k]) : 0.0f;
        dFh[i] = (_Float16)v;
    }
}

// ---------------------------------------------------------------------------
// K2: sd[b,c,t] = sum_k dF[c,k] * x[b,t+k]  via WMMA implicit GEMM
//     (channel diff folded into the filters). M=64ch, K=256, N=16384 time.
//     1024 blocks: (b, 256-wide time chunk). 8 waves x 8 tiles x 8 K-steps.
// ---------------------------------------------------------------------------
__global__ void conv_diff_wmma(const float* __restrict__ x,     // [16,16633]
                               const _Float16* __restrict__ dFh,// [64,256]
                               float* __restrict__ sd)          // [16,64,16384]
{
    __shared__ _Float16 sA[N1 * KFP];   // 32 KB diffed filters (f16)
    __shared__ float    sx[512];        // x window for this time chunk

    int b    = blockIdx.x >> 6;
    int tc   = blockIdx.x & 63;
    int t0   = tc * 256;
    int tid  = threadIdx.x;
    int wave = tid >> 5;
    int lane = tid & 31;

    for (int i = tid; i < N1 * KFP; i += 256) sA[i] = dFh[i];
    for (int i = tid; i < 512; i += 256) {
        int gi = t0 + i;
        sx[i] = (gi < TLEN) ? x[b * TLEN + gi] : 0.0f;
    }
    __syncthreads();

    int arow = lane & 15;   // A row within tile (M)
    int coln = lane & 15;   // B/C column (N)
    int hi   = lane >> 4;   // lane-half selector

    for (int tile = wave; tile < 64; tile += 8) {
        int ct = tile >> 4;            // channel tile (0..3)
        int tt = tile & 15;            // time sub-tile (0..15)
        v8f acc = {};
        int chbase = ct * 16 + arow;
        int tbase  = tt * 16 + coln;
#pragma unroll
        for (int kk = 0; kk < KFP; kk += 32) {
            v16h a, bm;
#pragma unroll
            for (int e = 0; e < 16; ++e) {
                // 16-bit A 16x32 layout: K = 16*(e>>3) + 8*(lane>>4) + (e&7)
                int ka = kk + ((e >> 3) << 4) + (hi << 3) + (e & 7);
                a[e] = sA[chbase * KFP + ka];
                // 16-bit B 32x16 layout: K = 16*(e>>3) + 8*(lane>>4) + (e&7)
                int kb = kk + ((e >> 3) << 4) + (hi << 3) + (e & 7);
                bm[e] = (_Float16)sx[tbase + kb];
            }
            acc = __builtin_amdgcn_wmma_f32_16x16x32_f16(false, a, false, bm,
                                                         (short)0, acc, false, false);
        }
        // C layout: VGPR r -> row r + 8*(lane>>4); col = lane&15
#pragma unroll
        for (int r = 0; r < 8; ++r) {
            int ch = ct * 16 + r + (hi << 3);
            int t  = t0 + tt * 16 + coln;
            sd[(b * N1 + ch) * T1 + t] = acc[r];
        }
    }
}

// ---------------------------------------------------------------------------
// K3: depthwise Gabor FIR via banded-Toeplitz WMMA.
//     A[m,j] = w[j-m] for 0<=j-m<128 (16x160 band, 5 K-steps of 32),
//     B[j,n] = padded[t0 + 16n + j]  ->  C[m,n] = filtout[t0 + 16n + m].
//     A tiles live in registers and are reused across all 16384 outputs.
//     Qin = mean|sd| is fused into the staging loop (single sd read pass),
//     counting each sample exactly once. Also emits prevFrame_out.
// ---------------------------------------------------------------------------
__global__ void gabor_fir_wmma(const float* __restrict__ sd,        // [16,64,16384]
                               const float* __restrict__ prevFrame, // [16,64,127]
                               const float* __restrict__ gab,       // [16,64,128]
                               float* __restrict__ filtout,         // [16,64,16384]
                               float* __restrict__ prevOut,         // [16,64,127]
                               float* __restrict__ Qin)             // [16,64]
{
    __shared__ _Float16 sAh[16 * 160];
    __shared__ float    sw[KG];
    __shared__ float    spad[2048 + 160];
    __shared__ float    sred[256];

    int bn   = blockIdx.x;              // (b,n) pair, 0..1023
    int tid  = threadIdx.x;
    int wave = tid >> 5;
    int lane = tid & 31;
    int hi   = lane >> 4;

    if (tid < KG) sw[tid] = gab[bn * KG + tid];
    __syncthreads();

    for (int i = tid; i < 16 * 160; i += 256) {
        int m = i / 160, j = i - m * 160;
        int d = j - m;
        sAh[i] = (d >= 0 && d < KG) ? (_Float16)sw[d] : (_Float16)0.0f;
    }

    // prevFrame_out = sd[:, :, T1-127:]
    if (tid < PREV) prevOut[bn * PREV + tid] = sd[bn * T1 + (T1 - PREV) + tid];
    __syncthreads();

    // cache the banded A tiles in registers (reused for every time chunk)
    v16h Areg[5];
    int mrow = lane & 15;
#pragma unroll
    for (int kk5 = 0; kk5 < 5; ++kk5) {
        v16h a;
#pragma unroll
        for (int e = 0; e < 16; ++e) {
            int ka = kk5 * 32 + ((e >> 3) << 4) + (hi << 3) + (e & 7);
            a[e] = sAh[mrow * 160 + ka];
        }
        Areg[kk5] = a;
    }

    int coln = lane & 15;
    float qacc = 0.0f;                  // fused |sd| accumulation
    for (int chunk = 0; chunk < T1; chunk += 2048) {
        __syncthreads();
        // stage padded[chunk .. chunk+2207] : j<127 -> prevFrame, else sd[j-127]
        for (int i = tid; i < 2048 + 160; i += 256) {
            int j = chunk + i;
            float v;
            if (j < PREV) v = prevFrame[bn * PREV + j];
            else {
                int t = j - PREV;
                v = (t < T1) ? sd[bn * T1 + t] : 0.0f;
                // count each sd sample exactly once across chunks:
                // i<2048 covers j=0..16383; the last chunk's overlap region
                // (i>=2048, t<T1) covers the final 127-sample tail.
                if (t < T1 && (i < 2048 || chunk == T1 - 2048)) qacc += fabsf(v);
            }
            spad[i] = v;
        }
        __syncthreads();

        int base = wave * 256;   // this wave's 256 outputs within the chunk
        v8f acc = {};
#pragma unroll
        for (int kk5 = 0; kk5 < 5; ++kk5) {
            v16h bm;
#pragma unroll
            for (int e = 0; e < 16; ++e) {
                int kb = kk5 * 32 + ((e >> 3) << 4) + (hi << 3) + (e & 7);
                bm[e] = (_Float16)spad[base + coln * 16 + kb];
            }
            acc = __builtin_amdgcn_wmma_f32_16x16x32_f16(false, Areg[kk5], false, bm,
                                                         (short)0, acc, false, false);
        }
#pragma unroll
        for (int r = 0; r < 8; ++r) {
            int m = r + (hi << 3);
            int t = chunk + base + coln * 16 + m;
            filtout[bn * T1 + t] = acc[r];
        }
    }

    // block-reduce Qin
    __syncthreads();
    sred[tid] = qacc;
    __syncthreads();
    for (int off = 128; off > 0; off >>= 1) {
        if (tid < off) sred[tid] += sred[tid + off];
        __syncthreads();
    }
    if (tid == 0) Qin[bn] = sred[0] * (1.0f / (float)T1);
}

// ---------------------------------------------------------------------------
// K4: rfft(16384) per channel via packed 8192-pt complex radix-2 FFT in LDS,
//     LDS-resident twiddle table (one 8192nd-root table serves all stages),
//     then unpack + |X| * mask, fused masked spectral-centroid reduction.
//     fm[b,n] = | sum(mag*freq)/sum(mag) - fc[b,n] |
//     Dynamic LDS: 2*8192 (data) + 2*4096 (twiddles) floats = 96 KB.
// ---------------------------------------------------------------------------
__global__ void fft_centroid(const float* __restrict__ filtout, // [16,64,16384]
                             const float* __restrict__ mask,    // [64,8193]
                             const float* __restrict__ freqs,   // [8193]
                             const float* __restrict__ fcT,     // [16,64]
                             float* __restrict__ fm)            // [16,64]
{
    extern __shared__ float smem[];
    float* re   = smem;               // [8192]
    float* im   = smem + M2;          // [8192]
    float* tw_r = smem + 2 * M2;      // [4096]
    float* tw_i = tw_r + (M2 >> 1);   // [4096]

    int bn  = blockIdx.x;
    int n   = bn & 63;
    int tid = threadIdx.x;
    const float* xin = filtout + bn * T1;

    // twiddle table: tw[j] = e^{-2*pi*i*j/8192}, j = 0..4095
    for (int j = tid; j < (M2 >> 1); j += 256) {
        float c, s;
        __sincosf(-PI_F * (float)j / 4096.0f, &s, &c);
        tw_r[j] = c; tw_i[j] = s;
    }
    // load even/odd packed, bit-reversed (13 bits)
    for (int j = tid; j < M2; j += 256) {
        unsigned r = __brev((unsigned)j) >> 19;
        re[r] = xin[2 * j];
        im[r] = xin[2 * j + 1];
    }
    __syncthreads();

    // 13 radix-2 DIT stages; stage twiddle = tw[pos << (12-st)]
    for (int st = 0; st < 13; ++st) {
        int half = 1 << st;
        int shft = 12 - st;
        for (int bf = tid; bf < (M2 >> 1); bf += 256) {
            int pos = bf & (half - 1);
            int i0  = ((bf >> st) << (st + 1)) + pos;
            int i1  = i0 + half;
            int jt  = pos << shft;
            float c = tw_r[jt], sn = tw_i[jt];
            float xr = re[i1], xi = im[i1];
            float tr = xr * c - xi * sn;
            float ti = xr * sn + xi * c;
            float ur = re[i0], ui = im[i0];
            re[i0] = ur + tr; im[i0] = ui + ti;
            re[i1] = ur - tr; im[i1] = ui - ti;
        }
        __syncthreads();
    }

    // unpack real-input spectrum, magnitude, masked centroid accumulation
    float s0 = 0.0f, s1 = 0.0f;
    const float* mrow = mask + n * FB;
    for (int k = tid; k <= M2; k += 256) {
        int k1 = k & (M2 - 1);
        int k2 = (M2 - k) & (M2 - 1);
        float zr = re[k1], zi = im[k1];
        float wr = re[k2], wi = im[k2];
        float Er = 0.5f * (zr + wr), Ei = 0.5f * (zi - wi);
        float tr = 0.5f * (zr - wr), ti = 0.5f * (zi + wi);
        float Or = ti, Oi = -tr;                    // O = -i*(tr + i*ti)
        float c, sn;
        __sincosf(-PI_F * (float)k / (float)M2, &sn, &c); // e^{-2*pi*i*k/16384}
        float Xr = Er + c * Or - sn * Oi;
        float Xi = Ei + c * Oi + sn * Or;
        float mag = sqrtf(Xr * Xr + Xi * Xi) * mrow[k];
        s0 += mag;
        s1 += mag * freqs[k];
    }
    __syncthreads();           // everyone done reading re/im before reuse
    re[tid] = s0; im[tid] = s1;
    __syncthreads();
    for (int off = 128; off > 0; off >>= 1) {
        if (tid < off) { re[tid] += re[tid + off]; im[tid] += im[tid + off]; }
        __syncthreads();
    }
    if (tid == 0) fm[bn] = fabsf(im[0] / re[0] - fcT[bn]);
}

// ---------------------------------------------------------------------------
// K5: batchnorm over batch + 3 small 16x64x64 matmuls + tanh head (f32 VALU)
// ---------------------------------------------------------------------------
__global__ void head_kernel(const float* __restrict__ fm,   // [16,64]
                            const float* __restrict__ Qin,  // [16,64]
                            const float* __restrict__ bnw, const float* __restrict__ bnb,
                            const float* __restrict__ Wh,  const float* __restrict__ bh,
                            const float* __restrict__ Wfm, const float* __restrict__ bfm,
                            const float* __restrict__ Wq,  const float* __restrict__ bq,
                            float* __restrict__ fm_bn_out,  // [16,64]
                            float* __restrict__ Qhid_out)   // [16,64]
{
    __shared__ float s_fmbn[BS * N1];
    __shared__ float s_h[BS * N1];
    __shared__ float mu[N1], iv[N1];
    int tid = threadIdx.x;

    if (tid < N1) {
        float m = 0.0f;
        for (int b = 0; b < BS; ++b) m += fm[b * N1 + tid];
        m *= (1.0f / BS);
        float v = 0.0f;
        for (int b = 0; b < BS; ++b) {
            float d = fm[b * N1 + tid] - m;
            v += d * d;
        }
        v *= (1.0f / BS);
        mu[tid] = m;
        iv[tid] = rsqrtf(v + 1e-5f);
    }
    __syncthreads();

    for (int i = tid; i < BS * N1; i += 256) {
        int nn = i & 63;
        float x = (fm[i] - mu[nn]) * iv[nn] * bnw[nn] + bnb[nn];
        s_fmbn[i]    = x;
        fm_bn_out[i] = x;
    }
    __syncthreads();

    for (int i = tid; i < BS * N1; i += 256) {
        int b = i >> 6, o = i & 63;
        float acc = bh[o];
        for (int k = 0; k < N1; ++k) acc += s_fmbn[b * N1 + k] * Wh[o * N1 + k];
        s_h[i] = fmaxf(acc, 0.0f);
    }
    __syncthreads();

    for (int i = tid; i < BS * N1; i += 256) {
        int b = i >> 6, o = i & 63;
        float a1 = bfm[o], a2 = bq[o];
        for (int k = 0; k < N1; ++k) {
            a1 += s_h[b * N1 + k] * Wfm[o * N1 + k];
            a2 += Qin[b * N1 + k] * Wq[o * N1 + k];
        }
        float hm = 1.5f * tanhf(a1) + 0.5f;
        Qhid_out[i] = 2.0f * tanhf(a2) + 1.0f + hm;
    }
}

// ---------------------------------------------------------------------------
extern "C" void kernel_launch(void* const* d_in, const int* in_sizes, int n_in,
                              void* d_out, int out_size, void* d_ws, size_t ws_size,
                              hipStream_t stream) {
    const float* xIn          = (const float*)d_in[0];
    const float* QPrev        = (const float*)d_in[2];
    const float* prevFrame    = (const float*)d_in[3];
    const float* filtersFixed = (const float*)d_in[5];
    const float* FreqScale1   = (const float*)d_in[6];
    const float* fc_variable  = (const float*)d_in[7];
    const float* fcTiled      = (const float*)d_in[8];
    const float* filter_mask  = (const float*)d_in[9];
    const float* Qout_w       = (const float*)d_in[10];
    const float* Qout_b       = (const float*)d_in[11];
    const float* hid_w        = (const float*)d_in[12];
    const float* hid_b        = (const float*)d_in[13];
    const float* QFM_w        = (const float*)d_in[14];
    const float* QFM_b        = (const float*)d_in[15];
    const float* bn_w         = (const float*)d_in[16];
    const float* bn_b         = (const float*)d_in[17];

    float* out      = (float*)d_out;
    float* filtout  = out;                                  // 16*64*16384
    float* prevOut  = out + (size_t)BS * N1 * T1;           // 16*64*127
    float* QhidOut  = prevOut + (size_t)BS * N1 * PREV;     // 16*64
    float* fmbnOut  = QhidOut + BS * N1;                    // 16*64

    // workspace layout
    char* ws = (char*)d_ws;
    float*    ws_sd  = (float*)ws;                           // 64 MB
    float*    ws_gab = (float*)(ws + (size_t)BS * N1 * T1 * 4);
    _Float16* ws_dFh = (_Float16*)((char*)ws_gab + (size_t)BS * N1 * KG * 4);
    float*    ws_Qin = (float*)((char*)ws_dFh + (size_t)N1 * KFP * 2);
    float*    ws_fm  = ws_Qin + BS * N1;

    prep_kernel<<<(BS * N1 * KG) / 256, 256, 0, stream>>>(QPrev, fc_variable,
                                                          filtersFixed, ws_gab, ws_dFh);

    conv_diff_wmma<<<BS * (T1 / 256), 256, 0, stream>>>(xIn, ws_dFh, ws_sd);

    gabor_fir_wmma<<<BS * N1, 256, 0, stream>>>(ws_sd, prevFrame, ws_gab,
                                                filtout, prevOut, ws_Qin);

    fft_centroid<<<BS * N1, 256,
                   (2 * M2 + M2) * sizeof(float),   // 8192*2 data + 4096*2 twiddles
                   stream>>>(filtout, filter_mask, FreqScale1, fcTiled, ws_fm);

    head_kernel<<<1, 256, 0, stream>>>(ws_fm, ws_Qin, bn_w, bn_b,
                                       hid_w, hid_b, QFM_w, QFM_b, Qout_w, Qout_b,
                                       fmbnOut, QhidOut);
}
